// CoGNNLayer_47605417509003
// MI455X (gfx1250) — matrix operand order repaired and checked by reading, not compiled, not run on hardware
//
#include <hip/hip_runtime.h>
#include <hip/hip_bf16.h>

typedef __attribute__((ext_vector_type(16))) _Float16 v16h;
typedef __attribute__((ext_vector_type(8)))  _Float16 v8h;
typedef __attribute__((ext_vector_type(8)))  float    v8f;

#define DD 128
#define HH 16

// ---------------------------------------------------------------- utilities
__global__ void k_zero(float* __restrict__ p, long n) {
    long t = (long)blockIdx.x * blockDim.x + threadIdx.x;
    if (t < n) p[t] = 0.0f;
}

// layer_norm over D=128, one wave (32 lanes) per row, 4 elems/lane -> f16 out
__global__ void k_ln(const float* __restrict__ x, const float* __restrict__ g,
                     const float* __restrict__ b, _Float16* __restrict__ xh, int N) {
    int row  = blockIdx.x * (blockDim.x >> 5) + (threadIdx.x >> 5);
    int lane = threadIdx.x & 31;
    if (row >= N) return;
    const float* xr = x + (size_t)row * DD;
    float v[4], s = 0.0f, ss = 0.0f;
#pragma unroll
    for (int i = 0; i < 4; i++) { v[i] = xr[lane * 4 + i]; s += v[i]; ss += v[i] * v[i]; }
#pragma unroll
    for (int m = 16; m >= 1; m >>= 1) { s += __shfl_xor(s, m, 32); ss += __shfl_xor(ss, m, 32); }
    float mu  = s * (1.0f / 128.0f);
    float var = ss * (1.0f / 128.0f) - mu * mu;
    float inv = rsqrtf(var + 1e-5f);
#pragma unroll
    for (int i = 0; i < 4; i++) {
        int j = lane * 4 + i;
        xh[(size_t)row * DD + j] = (_Float16)((v[i] - mu) * inv * g[j] + b[j]);
    }
}

// w[K][Nc] (f32, row-major) -> wt[Nc][K] (f16)
__global__ void k_tr_f16(const float* __restrict__ w, _Float16* __restrict__ wt, int K, int Nc) {
    int t = blockIdx.x * blockDim.x + threadIdx.x;
    if (t >= K * Nc) return;
    int n = t / K, k = t % K;
    wt[t] = (_Float16)w[k * Nc + n];
}

// ep_w[16][128] -> wt[128][32] transposed + K-padded with zeros
__global__ void k_tr_ep(const float* __restrict__ w, _Float16* __restrict__ wt) {
    int t = blockIdx.x * blockDim.x + threadIdx.x;
    if (t >= 128 * 32) return;
    int n = t / 32, k = t % 32;
    wt[t] = (k < 16) ? (_Float16)w[k * 128 + n] : (_Float16)0.0f;
}

__global__ void k_deg(const int* __restrict__ ei, float* __restrict__ deg, int E) {
    int t = blockIdx.x * blockDim.x + threadIdx.x;
    if (t < E) atomicAdd(deg + ei[E + t], 1.0f);     // dst row of edge_index
}

__global__ void k_rsqrt(const float* __restrict__ deg, float* __restrict__ dinv, int N) {
    int t = blockIdx.x * blockDim.x + threadIdx.x;
    if (t < N) dinv[t] = rsqrtf(deg[t] + 1.0f);
}

// ---------------------------------------------------------------- WMMA tile loads
__device__ inline v16h load_a16(const _Float16* __restrict__ base, int lda, int m0, int k0, int lane) {
    int m = m0 + (lane & 15), half = lane >> 4;
    const _Float16* p = base + (size_t)m * lda + k0 + 8 * half;
    v8h a0 = *(const v8h*)p;            // K = k0 + 8*half + [0..7]
    v8h a1 = *(const v8h*)(p + 16);     // K = k0 + 16 + 8*half + [0..7]
    v16h a;
#pragma unroll
    for (int i = 0; i < 8; i++) { a[i] = a0[i]; a[8 + i] = a1[i]; }
    return a;
}
__device__ inline v16h load_b16(const _Float16* __restrict__ wt, int ldk, int n0, int k0, int lane) {
    int n = n0 + (lane & 15), half = lane >> 4;
    const _Float16* p = wt + (size_t)n * ldk + k0 + 8 * half;
    v8h b0 = *(const v8h*)p;            // K = k0 + 8*half + [0..7]
    v8h b1 = *(const v8h*)(p + 16);     // K = k0 + 16 + 8*half + [0..7]
    v16h b;
#pragma unroll
    for (int i = 0; i < 8; i++) { b[i] = b0[i]; b[8 + i] = b1[i]; }
    return b;
}

// xw = xn_h @ conv_w : [N,128] x [128,128], one wave per 16x16 output tile
__global__ void k_xwconv(const _Float16* __restrict__ xh, const _Float16* __restrict__ wt,
                         float* __restrict__ xw, int N) {
    int tile = blockIdx.x;
    int wave = threadIdx.x >> 5, lane = threadIdx.x & 31;
    int m0 = tile * 16, n0 = wave * 16;
    if (m0 >= N) return;
    v8f c = {};
#pragma unroll
    for (int k0 = 0; k0 < 128; k0 += 32) {
        v16h a = load_a16(xh, 128, m0, k0, lane);
        v16h b = load_b16(wt, 128, n0, k0, lane);
        c = __builtin_amdgcn_wmma_f32_16x16x32_f16(false, a, false, b, (short)0, c, false, false);
    }
    int half = lane >> 4, n = lane & 15;
#pragma unroll
    for (int r = 0; r < 8; r++)
        xw[(size_t)(m0 + r + 8 * half) * 128 + n0 + n] = c[r];
}

// action layer 1 pre-aggregation: xw1 = xn_h @ {ia1_w | oa1_w} : [N,128] x [128,16]
__global__ void k_xw1(const _Float16* __restrict__ xh, const _Float16* __restrict__ wt_in,
                      const _Float16* __restrict__ wt_out, float* __restrict__ xw_in,
                      float* __restrict__ xw_out, int N) {
    int tile = blockIdx.x, net = blockIdx.y;
    int lane = threadIdx.x & 31;
    int m0 = tile * 16;
    if (m0 >= N) return;
    const _Float16* wt = net ? wt_out : wt_in;
    float* xw = net ? xw_out : xw_in;
    v8f c = {};
#pragma unroll
    for (int k0 = 0; k0 < 128; k0 += 32) {
        v16h a = load_a16(xh, 128, m0, k0, lane);
        v16h b = load_b16(wt, 128, 0, k0, lane);
        c = __builtin_amdgcn_wmma_f32_16x16x32_f16(false, a, false, b, (short)0, c, false, false);
    }
    int half = lane >> 4, n = lane & 15;
#pragma unroll
    for (int r = 0; r < 8; r++)
        xw[(size_t)(m0 + r + 8 * half) * HH + n] = c[r];
}

// fused: edge_features = relu(edge_attr @ ep_w + ep_b); seg_sum into d_out[dst]
__global__ void k_edgefeat(const float* __restrict__ ea, const _Float16* __restrict__ ep_wt,
                           const float* __restrict__ ep_b, const int* __restrict__ ei,
                           float* __restrict__ out, int E) {
    int tile = blockIdx.x * (blockDim.x >> 5) + (threadIdx.x >> 5);
    int lane = threadIdx.x & 31;
    int e0 = tile * 16;
    if (e0 >= E) return;
    int m = lane & 15, half = lane >> 4, n = lane & 15;
    // A tile: 16 edges x K32 (K>=16 zero-padded)
    const float* row = ea + (size_t)(e0 + m) * 16 + 8 * half;
    v16h a;
#pragma unroll
    for (int i = 0; i < 8; i++) a[i] = (_Float16)row[i];
#pragma unroll
    for (int i = 8; i < 16; i++) a[i] = (_Float16)0.0f;
#pragma unroll
    for (int ct = 0; ct < 8; ct++) {
        const _Float16* bp = ep_wt + (size_t)(ct * 16 + n) * 32 + 8 * half;
        v8h b0 = *(const v8h*)bp, b1 = *(const v8h*)(bp + 16);
        v16h b;
#pragma unroll
        for (int i = 0; i < 8; i++) { b[i] = b0[i]; b[8 + i] = b1[i]; }
        v8f c = {};
        c = __builtin_amdgcn_wmma_f32_16x16x32_f16(false, a, false, b, (short)0, c, false, false);
        float bias = ep_b[ct * 16 + n];
#pragma unroll
        for (int r = 0; r < 8; r++) {
            int e = e0 + r + 8 * half;
            if (e < E) {
                float v = fmaxf(c[r] + bias, 0.0f);
                atomicAdd(out + (size_t)ei[E + e] * 128 + ct * 16 + n, v);
            }
        }
    }
}

// ---------------------------------------------------------------- graph passes
__global__ void k_agg1(const int* __restrict__ ei, const float* __restrict__ dinv1,
                       const float* __restrict__ xw_in, const float* __restrict__ xw_out,
                       float* __restrict__ agg_in, float* __restrict__ agg_out, int E) {
    int t = blockIdx.x * blockDim.x + threadIdx.x;
    int e = t >> 4, j = t & 15;
    if (e >= E) return;
    int s = ei[e], d = ei[E + e];
    float norm = dinv1[s] * dinv1[d];
    atomicAdd(agg_in  + (size_t)d * HH + j, xw_in [(size_t)s * HH + j] * norm);
    atomicAdd(agg_out + (size_t)d * HH + j, xw_out[(size_t)s * HH + j] * norm);
}

__global__ void k_h(const float* __restrict__ agg_in, const float* __restrict__ agg_out,
                    const float* __restrict__ xw_in, const float* __restrict__ xw_out,
                    const float* __restrict__ dinv1, const float* __restrict__ b_in,
                    const float* __restrict__ b_out, float* __restrict__ h_in,
                    float* __restrict__ h_out, int N) {
    int t = blockIdx.x * blockDim.x + threadIdx.x;
    if (t >= N * HH) return;
    int n = t >> 4, j = t & 15;
    float di = dinv1[n], sl = di * di;
    h_in[t]  = fmaxf(agg_in[t]  + xw_in[t]  * sl + b_in[j],  0.0f);
    h_out[t] = fmaxf(agg_out[t] + xw_out[t] * sl + b_out[j], 0.0f);
}

__global__ void k_xw2(const float* __restrict__ h_in, const float* __restrict__ h_out,
                      const float* __restrict__ w_in, const float* __restrict__ w_out,
                      float* __restrict__ xw2_in, float* __restrict__ xw2_out, int N) {
    int n = blockIdx.x * blockDim.x + threadIdx.x;
    if (n >= N) return;
    float a0 = 0, a1 = 0, c0 = 0, c1 = 0;
#pragma unroll
    for (int j = 0; j < HH; j++) {
        float hi = h_in[(size_t)n * HH + j], ho = h_out[(size_t)n * HH + j];
        a0 += hi * w_in[j * 2];  a1 += hi * w_in[j * 2 + 1];
        c0 += ho * w_out[j * 2]; c1 += ho * w_out[j * 2 + 1];
    }
    xw2_in[n * 2] = a0;  xw2_in[n * 2 + 1] = a1;
    xw2_out[n * 2] = c0; xw2_out[n * 2 + 1] = c1;
}

__global__ void k_agg2(const int* __restrict__ ei, const float* __restrict__ dinv1,
                       const float* __restrict__ xw2_in, const float* __restrict__ xw2_out,
                       float* __restrict__ lg_in, float* __restrict__ lg_out, int E) {
    int t = blockIdx.x * blockDim.x + threadIdx.x;
    int e = t >> 2, idx = t & 3;
    if (e >= E) return;
    int net = idx >> 1, c = idx & 1;
    int s = ei[e], d = ei[E + e];
    float norm = dinv1[s] * dinv1[d];
    const float* xw2 = net ? xw2_out : xw2_in;
    float* lg = net ? lg_out : lg_in;
    atomicAdd(lg + (size_t)d * 2 + c, xw2[(size_t)s * 2 + c] * norm);
}

__device__ inline float gumb(float u) { return -logf(-logf(u + 1e-10f) + 1e-10f); }

// forward value of gumbel_hard is exactly the one-hot; argmax((l+g)/tau)==argmax(l+g)
__global__ void k_flags(const float* __restrict__ lg_in, const float* __restrict__ lg_out,
                        const float* __restrict__ xw2_in, const float* __restrict__ xw2_out,
                        const float* __restrict__ dinv1, const float* __restrict__ bi,
                        const float* __restrict__ bo, const float* __restrict__ u_in,
                        const float* __restrict__ u_out, float* __restrict__ fin,
                        float* __restrict__ fout, int N) {
    int n = blockIdx.x * blockDim.x + threadIdx.x;
    if (n >= N) return;
    float sl = dinv1[n] * dinv1[n];
    float z0 = lg_in[n * 2]     + xw2_in[n * 2]     * sl + bi[0] + gumb(u_in[n * 2]);
    float z1 = lg_in[n * 2 + 1] + xw2_in[n * 2 + 1] * sl + bi[1] + gumb(u_in[n * 2 + 1]);
    fin[n] = (z0 >= z1) ? 1.0f : 0.0f;
    float y0 = lg_out[n * 2]     + xw2_out[n * 2]     * sl + bo[0] + gumb(u_out[n * 2]);
    float y1 = lg_out[n * 2 + 1] + xw2_out[n * 2 + 1] * sl + bo[1] + gumb(u_out[n * 2 + 1]);
    fout[n] = (y0 >= y1) ? 1.0f : 0.0f;
}

__global__ void k_ew(const int* __restrict__ ei, const float* __restrict__ fin,
                     const float* __restrict__ fout, float* __restrict__ ew,
                     float* __restrict__ deg_ew, int E) {
    int e = blockIdx.x * blockDim.x + threadIdx.x;
    if (e >= E) return;
    int s = ei[e], d = ei[E + e];
    float w = fin[d] * fout[s];
    ew[e] = w;
    if (w != 0.0f) atomicAdd(deg_ew + d, w);
}

// main scatter: one wave per edge, 128 floats; wave-uniform skip of dead edges
__global__ void k_scatter(const int* __restrict__ ei, const float* __restrict__ ew,
                          const float* __restrict__ dinv2, const float* __restrict__ xw,
                          float* __restrict__ out, int E) {
    int e = blockIdx.x * (blockDim.x >> 5) + (threadIdx.x >> 5);
    if (e >= E) return;
    float w = ew[e];
    if (w == 0.0f) return;
    int lane = threadIdx.x & 31;
    int s = ei[e], d = ei[E + e];
    float norm = dinv2[s] * w * dinv2[d];
    const float4 v = *(const float4*)(xw + (size_t)s * 128 + lane * 4);
    float* o = out + (size_t)d * 128 + lane * 4;
    atomicAdd(o,     v.x * norm);
    atomicAdd(o + 1, v.y * norm);
    atomicAdd(o + 2, v.z * norm);
    atomicAdd(o + 3, v.w * norm);
}

__global__ void k_final(const float* __restrict__ xw, const float* __restrict__ dinv2,
                        const float* __restrict__ cb, float* __restrict__ out, int N) {
    int t = blockIdx.x * blockDim.x + threadIdx.x;
    if (t >= N * 128) return;
    int n = t >> 7, j = t & 127;
    float di = dinv2[n];
    out[t] = fmaxf(out[t] + xw[t] * di * di + cb[j], 0.0f);
}

// ---------------------------------------------------------------- launcher
extern "C" void kernel_launch(void* const* d_in, const int* in_sizes, int n_in,
                              void* d_out, int out_size, void* d_ws, size_t ws_size,
                              hipStream_t stream) {
    const float* x        = (const float*)d_in[0];
    const int*   ei       = (const int*)  d_in[1];
    const float* ea       = (const float*)d_in[2];
    const float* u_in     = (const float*)d_in[3];
    const float* u_out    = (const float*)d_in[4];
    const float* ln_g     = (const float*)d_in[5];
    const float* ln_b     = (const float*)d_in[6];
    const float* conv_w   = (const float*)d_in[7];
    const float* conv_b   = (const float*)d_in[8];
    const float* ep_w     = (const float*)d_in[9];
    const float* ep_b     = (const float*)d_in[10];
    const float* ia1_w    = (const float*)d_in[11];
    const float* ia1_b    = (const float*)d_in[12];
    const float* ia2_w    = (const float*)d_in[13];
    const float* ia2_b    = (const float*)d_in[14];
    const float* oa1_w    = (const float*)d_in[15];
    const float* oa1_b    = (const float*)d_in[16];
    const float* oa2_w    = (const float*)d_in[17];
    const float* oa2_b    = (const float*)d_in[18];
    float* out = (float*)d_out;

    const int N = in_sizes[0] / DD;
    const int E = in_sizes[1] / 2;

    // ---- workspace carve-out (256B aligned chunks)
    char* ws = (char*)d_ws;
    size_t off = 0;
    auto alloc = [&](size_t bytes) -> char* {
        char* p = ws + off;
        off = (off + bytes + 255) & ~(size_t)255;
        return p;
    };
    // contiguous accumulator region (zeroed with one launch): 38 floats/node
    float* accum   = (float*)alloc((size_t)N * 38 * sizeof(float));
    float* deg1    = accum;             // N
    float* deg_ew  = accum + (size_t)N; // N
    float* agg_in  = accum + (size_t)2  * N;  // 16N
    float* agg_out = accum + (size_t)18 * N;  // 16N
    float* lg_in   = accum + (size_t)34 * N;  // 2N
    float* lg_out  = accum + (size_t)36 * N;  // 2N

    _Float16* xn_h   = (_Float16*)alloc((size_t)N * DD * sizeof(_Float16));
    float* xw_conv   = (float*)alloc((size_t)N * DD * sizeof(float));
    float* xw_in     = (float*)alloc((size_t)N * HH * sizeof(float));
    float* xw_out    = (float*)alloc((size_t)N * HH * sizeof(float));
    float* h_in      = (float*)alloc((size_t)N * HH * sizeof(float));
    float* h_out     = (float*)alloc((size_t)N * HH * sizeof(float));
    float* xw2_in    = (float*)alloc((size_t)N * 2 * sizeof(float));
    float* xw2_out   = (float*)alloc((size_t)N * 2 * sizeof(float));
    float* dinv1     = (float*)alloc((size_t)N * sizeof(float));
    float* dinv2     = (float*)alloc((size_t)N * sizeof(float));
    float* fin       = (float*)alloc((size_t)N * sizeof(float));
    float* fout      = (float*)alloc((size_t)N * sizeof(float));
    float* ew        = (float*)alloc((size_t)E * sizeof(float));
    _Float16* conv_wt = (_Float16*)alloc((size_t)128 * 128 * sizeof(_Float16));
    _Float16* ia1_wt  = (_Float16*)alloc((size_t)16 * 128 * sizeof(_Float16));
    _Float16* oa1_wt  = (_Float16*)alloc((size_t)16 * 128 * sizeof(_Float16));
    _Float16* ep_wt   = (_Float16*)alloc((size_t)128 * 32 * sizeof(_Float16));

    const int B = 256;
    auto cdiv = [](long a, long b) { return (int)((a + b - 1) / b); };

    // 0) zero accumulators and output
    k_zero<<<cdiv((long)N * 38, B), B, 0, stream>>>(accum, (long)N * 38);
    k_zero<<<cdiv((long)N * DD, B), B, 0, stream>>>(out, (long)N * DD);

    // 1) layernorm -> f16
    k_ln<<<cdiv(N, 8), B, 0, stream>>>(x, ln_g, ln_b, xn_h, N);

    // 2) weight prep (transpose + f16)
    k_tr_f16<<<cdiv(128 * 128, B), B, 0, stream>>>(conv_w, conv_wt, 128, 128);
    k_tr_f16<<<cdiv(128 * 16, B), B, 0, stream>>>(ia1_w, ia1_wt, 128, 16);
    k_tr_f16<<<cdiv(128 * 16, B), B, 0, stream>>>(oa1_w, oa1_wt, 128, 16);
    k_tr_ep<<<cdiv(128 * 32, B), B, 0, stream>>>(ep_w, ep_wt);

    // 3) degree (ones weights) + dinv1
    k_deg<<<cdiv(E, B), B, 0, stream>>>(ei, deg1, E);
    k_rsqrt<<<cdiv(N, B), B, 0, stream>>>(deg1, dinv1, N);

    // 4) action nets, layer 1 (WMMA) + edge aggregation + relu
    k_xw1<<<dim3(cdiv(N, 16), 2), 32, 0, stream>>>(xn_h, ia1_wt, oa1_wt, xw_in, xw_out, N);
    k_agg1<<<cdiv((long)E * HH, B), B, 0, stream>>>(ei, dinv1, xw_in, xw_out, agg_in, agg_out, E);
    k_h<<<cdiv((long)N * HH, B), B, 0, stream>>>(agg_in, agg_out, xw_in, xw_out, dinv1,
                                                 ia1_b, oa1_b, h_in, h_out, N);

    // 5) action nets, layer 2 + gumbel-hard flags
    k_xw2<<<cdiv(N, B), B, 0, stream>>>(h_in, h_out, ia2_w, oa2_w, xw2_in, xw2_out, N);
    k_agg2<<<cdiv((long)E * 4, B), B, 0, stream>>>(ei, dinv1, xw2_in, xw2_out, lg_in, lg_out, E);
    k_flags<<<cdiv(N, B), B, 0, stream>>>(lg_in, lg_out, xw2_in, xw2_out, dinv1,
                                          ia2_b, oa2_b, u_in, u_out, fin, fout, N);

    // 6) edge weights (binary) + weighted degree + dinv2
    k_ew<<<cdiv(E, B), B, 0, stream>>>(ei, fin, fout, ew, deg_ew, E);
    k_rsqrt<<<cdiv(N, B), B, 0, stream>>>(deg_ew, dinv2, N);

    // 7) main conv GEMM (WMMA) + sparse scatter
    k_xwconv<<<cdiv(N, 16), B, 0, stream>>>(xn_h, conv_wt, xw_conv, N);
    k_scatter<<<cdiv(E, 8), B, 0, stream>>>(ei, ew, dinv2, xw_conv, out, E);

    // 8) fused edge-feature GEMM (WMMA) + scatter of relu(edge_features)
    k_edgefeat<<<cdiv(cdiv(E, 16), 8), B, 0, stream>>>(ea, ep_wt, ep_b, ei, out, E);

    // 9) self-loop + bias + relu
    k_final<<<cdiv((long)N * DD, B), B, 0, stream>>>(xw_conv, dinv2, conv_b, out, N);
}
